// HashingMemory_50869592654821
// MI455X (gfx1250) — matrix operand order repaired and checked by reading, not compiled, not run on hardware
//
#include <hip/hip_runtime.h>
#include <cstdint>
#include <cstddef>

#define HEADS 4
#define K_DIM 512
#define HALF 256
#define N_KEYS 512
#define KNN 32
#define V_DIM 1024
#define IN_DIM 1024
#define BS 4096
#define QCOLS (HEADS * K_DIM)      /* 2048 */
#define SCOLS (HEADS * 2 * N_KEYS) /* 4096 */

typedef __attribute__((ext_vector_type(16))) __bf16 v16bf;
typedef __attribute__((ext_vector_type(8)))  float  v8f;

#define NEG_INF (-3.0e38f)

__device__ __forceinline__ unsigned short f2bf(float f) {
  unsigned u = __float_as_uint(f);
  unsigned r = u + 0x7fffu + ((u >> 16) & 1u); // round-to-nearest-even
  return (unsigned short)(r >> 16);
}

// async global->LDS copy, 16B per lane, ASYNCcnt-tracked (CDNA5).
// SADDR form: uniform 64-bit base in SGPR pair + 32-bit per-lane byte offset.
__device__ __forceinline__ void async_copy_b128(unsigned long long sbase,
                                                unsigned voff_bytes,
                                                unsigned lds_addr) {
  asm volatile("global_load_async_to_lds_b128 %0, %1, %2"
               :: "v"(lds_addr), "v"(voff_bytes), "s"(sbase) : "memory");
}
__device__ __forceinline__ void wait_async0() {
  asm volatile("s_wait_asynccnt 0x0" ::: "memory");
}
__device__ __forceinline__ unsigned lds_lo32(const void* p) {
  return (unsigned)(uintptr_t)p;  // low 32 bits of flat LDS addr = LDS offset
}

// ---------------------------------------------------------------------------
// K0: convert x (f32) -> bf16, 4 elements/thread
// ---------------------------------------------------------------------------
__global__ void k_convert_x(const float* __restrict__ x,
                            unsigned short* __restrict__ xb) {
  int i = blockIdx.x * blockDim.x + threadIdx.x; // [0, BS*IN_DIM/4)
  float4 v = reinterpret_cast<const float4*>(x)[i];
  ushort4 o;
  o.x = f2bf(v.x); o.y = f2bf(v.y); o.z = f2bf(v.z); o.w = f2bf(v.w);
  reinterpret_cast<ushort4*>(xb)[i] = o;
}

// ---------------------------------------------------------------------------
// K1: W2T[col][i] = sum_d w_query[i][c2*256+d] * keys[c2*512+n][d]
//     col = c2*512 + n.  Stored transposed (K-contiguous) as bf16 so the big
//     GEMM's B operand stages into LDS with plain contiguous 16B copies.
// ---------------------------------------------------------------------------
__global__ void k_build_w2(const float* __restrict__ wq,
                           const float* __restrict__ keys,
                           unsigned short* __restrict__ w2bt) {
  __shared__ float wt[16][17];
  __shared__ float kt[16][17];
  const int tx = threadIdx.x, ty = threadIdx.y;
  const int i0   = blockIdx.x * 16;   // i tile   (IN_DIM)
  const int col0 = blockIdx.y * 16;   // col tile (SCOLS), 512-aligned blocks
  const int c2 = col0 >> 9;           // head*2+side, uniform within tile
  const int n0 = col0 & (N_KEYS - 1);
  float acc = 0.f;
  for (int d0 = 0; d0 < HALF; d0 += 16) {
    wt[ty][tx] = wq[(size_t)(i0 + ty) * QCOLS + c2 * HALF + d0 + tx];
    kt[ty][tx] = keys[(size_t)(c2 * N_KEYS + n0 + ty) * HALF + d0 + tx];
    __syncthreads();
#pragma unroll
    for (int dd = 0; dd < 16; ++dd)
      acc += wt[tx][dd] * kt[ty][dd];   // thread: (i = i0+tx, col = col0+ty)
    __syncthreads();
  }
  w2bt[(size_t)(col0 + ty) * IN_DIM + i0 + tx] = f2bf(acc);
}

// ---------------------------------------------------------------------------
// K2: scores = xb(4096x1024 bf16) @ W2T^T -> f32 (4096x4096)
//     128x128 block tile, 8 waves (2x4), wave tile 64x32 = 4x2 WMMA tiles,
//     K-step 32, v_wmma_f32_16x16x32_bf16.  Double-buffered LDS filled by
//     GLOBAL_LOAD_ASYNC_TO_LDS_B128 (SADDR form), one barrier per K-step,
//     fully unrolled so accumulators keep stable registers (no v_mov churn).
// ---------------------------------------------------------------------------
#define LDS_STRIDE 40  /* halfs; 80B row pitch, keeps 16B alignment, skews banks */
#define KSTEPS (IN_DIM / 32)

__global__ void __launch_bounds__(256) k_scores_gemm(
    const unsigned short* __restrict__ xb,    // BS x IN_DIM (K-contig)
    const unsigned short* __restrict__ w2bt,  // SCOLS x IN_DIM (K-contig)
    float* __restrict__ scores)               // BS x SCOLS
{
  __shared__ __align__(16) unsigned short As[2][128 * LDS_STRIDE];
  __shared__ __align__(16) unsigned short Bs[2][128 * LDS_STRIDE];

  const int t    = threadIdx.x;
  const int lane = t & 31;
  const int wave = t >> 5;
  const int wm = wave >> 2;          // 0..1
  const int wn = wave & 3;           // 0..3
  const int row0 = blockIdx.y * 128;
  const int col0 = blockIdx.x * 128;
  const int lr  = lane & 15;
  const int lh  = lane >> 4;         // lane-half select
  const int kbA = lh * 8;            // A: K runs of 8 at kbA and 16+kbA
  const int kbB = lh * 8;            // B: K runs of 8 at kbB and 16+kbB

  // this thread's two staging chunks (16B each per matrix)
  const int r0 = t >> 2,         ko0 = (t & 3) * 8;
  const int r1 = (t + 256) >> 2, ko1 = (t & 3) * 8;  // c1 = t+256, same ko

  const unsigned long long baseA = (unsigned long long)(uintptr_t)xb;
  const unsigned long long baseB = (unsigned long long)(uintptr_t)w2bt;
  // byte offsets of k=0 chunks (K dim is contiguous bf16 => 2 B/elem)
  const unsigned offA0 = (unsigned)(((row0 + r0) * IN_DIM + ko0) * 2);
  const unsigned offA1 = (unsigned)(((row0 + r1) * IN_DIM + ko1) * 2);
  const unsigned offB0 = (unsigned)(((col0 + r0) * IN_DIM + ko0) * 2);
  const unsigned offB1 = (unsigned)(((col0 + r1) * IN_DIM + ko1) * 2);
  // LDS destinations for both buffers
  unsigned ldsA0[2], ldsA1[2], ldsB0[2], ldsB1[2];
#pragma unroll
  for (int bufi = 0; bufi < 2; ++bufi) {
    ldsA0[bufi] = lds_lo32(&As[bufi][r0 * LDS_STRIDE + ko0]);
    ldsA1[bufi] = lds_lo32(&As[bufi][r1 * LDS_STRIDE + ko1]);
    ldsB0[bufi] = lds_lo32(&Bs[bufi][r0 * LDS_STRIDE + ko0]);
    ldsB1[bufi] = lds_lo32(&Bs[bufi][r1 * LDS_STRIDE + ko1]);
  }

  v8f acc[4][2];
#pragma unroll
  for (int a = 0; a < 4; ++a)
#pragma unroll
    for (int b = 0; b < 2; ++b) acc[a][b] = (v8f){};

  union Frag { uint4 q[2]; v16bf v; };

  // prologue: stage K-tile 0 into buffer 0
  async_copy_b128(baseA, offA0, ldsA0[0]);
  async_copy_b128(baseA, offA1, ldsA1[0]);
  async_copy_b128(baseB, offB0, ldsB0[0]);
  async_copy_b128(baseB, offB1, ldsB1[0]);

#pragma unroll
  for (int kk = 0; kk < KSTEPS; ++kk) {
    wait_async0();          // own prefetch into buf landed in LDS
    __syncthreads();        // all waves: writes visible, spare buf free

    const int buf = kk & 1;
    if (kk + 1 < KSTEPS) {  // prefetch next K-tile into spare buffer
      const unsigned kb = (unsigned)((kk + 1) * 32 * 2);  // byte offset step
      const int nb = buf ^ 1;
      async_copy_b128(baseA, offA0 + kb, ldsA0[nb]);
      async_copy_b128(baseA, offA1 + kb, ldsA1[nb]);
      async_copy_b128(baseB, offB0 + kb, ldsB0[nb]);
      async_copy_b128(baseB, offB1 + kb, ldsB1[nb]);
    }

    Frag af[4], bfr[2];
#pragma unroll
    for (int a = 0; a < 4; ++a) {
      const unsigned short* p = &As[buf][(wm * 64 + a * 16 + lr) * LDS_STRIDE + kbA];
      af[a].q[0] = *reinterpret_cast<const uint4*>(p);        // K = kbA..kbA+7
      af[a].q[1] = *reinterpret_cast<const uint4*>(p + 16);   // K = 16+kbA..
    }
#pragma unroll
    for (int b = 0; b < 2; ++b) {
      const unsigned short* p = &Bs[buf][(wn * 32 + b * 16 + lr) * LDS_STRIDE + kbB];
      bfr[b].q[0] = *reinterpret_cast<const uint4*>(p);       // K = kbB..kbB+7
      bfr[b].q[1] = *reinterpret_cast<const uint4*>(p + 16);  // K = 16+kbB..
    }
#pragma unroll
    for (int a = 0; a < 4; ++a)
#pragma unroll
      for (int b = 0; b < 2; ++b)
        acc[a][b] = __builtin_amdgcn_wmma_f32_16x16x32_bf16(
            false, af[a].v, false, bfr[b].v, (short)0, acc[a][b], false, false);
  }

  // D layout: VGPR r -> m = r + 8*(lane>=16), n = lane&15
#pragma unroll
  for (int a = 0; a < 4; ++a)
#pragma unroll
    for (int b = 0; b < 2; ++b)
#pragma unroll
      for (int r = 0; r < 8; ++r) {
        int gr = row0 + wm * 64 + a * 16 + r + 8 * lh;
        int gc = col0 + wn * 32 + b * 16 + lr;
        scores[(size_t)gr * SCOLS + gc] = acc[a][b][r];
      }
}

// ---------------------------------------------------------------------------
// K3: per (b,h) wave: top-32 per side, cartesian top-32, softmax.
//     All-register, shfl-based; 4 independent waves per 128-thread block.
// ---------------------------------------------------------------------------
__device__ __forceinline__ void wave_argmax(float& v, unsigned& aux) {
#pragma unroll
  for (int off = 16; off > 0; off >>= 1) {
    float    ov = __shfl_xor(v, off, 32);
    unsigned oa = __shfl_xor(aux, off, 32);
    if (ov > v || (ov == v && oa < aux)) { v = ov; aux = oa; }
  }
}

__global__ void __launch_bounds__(128) k_topk(const float* __restrict__ scores,
                                              float* __restrict__ wts,
                                              int* __restrict__ idxo) {
  const int lane = threadIdx.x & 31;
  const int pair = blockIdx.x * 4 + (threadIdx.x >> 5); // b*HEADS + h
  const int b = pair >> 2, h = pair & 3;

  float my_s1 = 0.f, my_s2 = 0.f;
  int   my_i1 = 0,   my_i2 = 0;

#pragma unroll
  for (int s = 0; s < 2; ++s) {
    const float* sp = scores + (size_t)b * SCOLS + (h * 2 + s) * N_KEYS;
    float vals[16];
#pragma unroll
    for (int j = 0; j < 16; ++j) vals[j] = sp[j * 32 + lane]; // n = j*32+lane
    unsigned cons = 0;
    for (int k = 0; k < KNN; ++k) {
      float bv = NEG_INF; int bj = 0;
#pragma unroll
      for (int j = 0; j < 16; ++j) {
        float xv = ((cons >> j) & 1u) ? NEG_INF : vals[j];
        if (xv > bv) { bv = xv; bj = j; }
      }
      float v = bv;
      unsigned aux = (unsigned)(bj * 32 + lane); // == n
      wave_argmax(v, aux);
      if (lane == (int)(aux & 31u)) cons |= 1u << (aux >> 5);
      if (lane == k) {
        if (s == 0) { my_s1 = v; my_i1 = (int)aux; }
        else        { my_s2 = v; my_i2 = (int)aux; }
      }
    }
  }

  // cartesian candidates: lane owns i = lane, cand[j] = s1[i] + s2[j]
  float cand[32];
#pragma unroll
  for (int j = 0; j < 32; ++j) cand[j] = my_s1 + __shfl(my_s2, j, 32);

  unsigned cons2 = 0;
  float out_v = NEG_INF; int out_idx = 0;
  for (int k = 0; k < KNN; ++k) {
    float bv = NEG_INF; int bj = 0;
#pragma unroll
    for (int j = 0; j < 32; ++j) {
      float xv = ((cons2 >> j) & 1u) ? NEG_INF : cand[j];
      if (xv > bv) { bv = xv; bj = j; }
    }
    float v = bv;
    unsigned aux = (unsigned)((lane << 5) | bj); // flat f = i*32 + j
    wave_argmax(v, aux);
    int wi = (int)(aux >> 5), wj = (int)(aux & 31u);
    if (lane == wi) cons2 |= 1u << wj;
    int sel1 = __shfl(my_i1, wi, 32);
    int sel2 = __shfl(my_i2, wj, 32);
    if (lane == k) { out_v = v; out_idx = sel1 * N_KEYS + sel2; }
  }

  // softmax over the 32 selected (descending -> lane 0 holds max)
  float m = __shfl(out_v, 0, 32);
  float e = __expf(out_v - m);
  float sum = e;
#pragma unroll
  for (int off = 16; off > 0; off >>= 1) sum += __shfl_xor(sum, off, 32);

  wts [(size_t)pair * KNN + lane] = e / sum;
  idxo[(size_t)pair * KNN + lane] = out_idx;
}

// ---------------------------------------------------------------------------
// K4: out[b,:] = sum over 128 (h,k) of w * values[idx]; one block per row.
//     HBM-bound stage: ~2.1 GB of gathered, intra-row-coalesced reads.
// ---------------------------------------------------------------------------
__global__ void __launch_bounds__(256) k_gather(
    const float* __restrict__ values,
    const float* __restrict__ wts, const int* __restrict__ idx,
    float* __restrict__ out) {
  __shared__ float sw[HEADS * KNN];
  __shared__ int   si[HEADS * KNN];
  const int b = blockIdx.x, t = threadIdx.x;
  if (t < HEADS * KNN) {
    sw[t] = wts[(size_t)b * HEADS * KNN + t];
    si[t] = idx[(size_t)b * HEADS * KNN + t];
  }
  __syncthreads();
  float4 acc = {0.f, 0.f, 0.f, 0.f};
#pragma unroll 4
  for (int r = 0; r < HEADS * KNN; ++r) {
    const float4 v =
        reinterpret_cast<const float4*>(values + (size_t)si[r] * V_DIM)[t];
    const float w = sw[r];
    acc.x += w * v.x; acc.y += w * v.y; acc.z += w * v.z; acc.w += w * v.w;
  }
  reinterpret_cast<float4*>(out + (size_t)b * V_DIM)[t] = acc;
}

// ---------------------------------------------------------------------------
extern "C" void kernel_launch(void* const* d_in, const int* in_sizes, int n_in,
                              void* d_out, int out_size, void* d_ws,
                              size_t ws_size, hipStream_t stream) {
  (void)in_sizes; (void)n_in; (void)out_size; (void)ws_size;
  const float* x      = (const float*)d_in[0];
  const float* wq     = (const float*)d_in[1];
  const float* keys   = (const float*)d_in[2];
  const float* values = (const float*)d_in[3];
  float* out = (float*)d_out;

  char* ws = (char*)d_ws;
  unsigned short* xb     = (unsigned short*)(ws);                       //  8 MB
  unsigned short* w2bt   = (unsigned short*)(ws + ((size_t)8  << 20));  //  8 MB
  float*          scores = (float*)         (ws + ((size_t)16 << 20));  // 64 MB
  float*          wts    = (float*)         (ws + ((size_t)80 << 20));  //  2 MB
  int*            idx    = (int*)           (ws + ((size_t)82 << 20));  //  2 MB

  k_convert_x<<<(BS * IN_DIM / 4) / 256, 256, 0, stream>>>(x, xb);

  dim3 g1(IN_DIM / 16, SCOLS / 16), b1(16, 16);
  k_build_w2<<<g1, b1, 0, stream>>>(wq, keys, w2bt);

  dim3 g2(SCOLS / 128, BS / 128);
  k_scores_gemm<<<g2, 256, 0, stream>>>(xb, w2bt, scores);

  k_topk<<<(BS * HEADS) / 4, 128, 0, stream>>>(scores, wts, idx);

  k_gather<<<BS, 256, 0, stream>>>(values, wts, idx, out);
}
